// parsingNet_60421599920390
// MI455X (gfx1250) — matrix operand order repaired and checked by reading, not compiled, not run on hardware
//
#include <hip/hip_runtime.h>
#include <cstdint>

typedef __attribute__((ext_vector_type(16))) _Float16 v16h;
typedef __attribute__((ext_vector_type(8)))  float    v8f;

#define H_G 9
#define W_G 25
#define N_NODES 225
#define BATCH 128

// ---------------------------------------------------------------------------
// Stage 1: 1x1 conv pool  x4[B,2048,225] * pool_w[8,2048] + pool_b -> fea[B,225,8]
// grid: (8 hw-tiles, 128 batches), block: 256 (8 waves). Each wave owns 32 hw
// positions and 256 channels; partials reduced through LDS. Coalesced over hw.
// ---------------------------------------------------------------------------
__global__ void pool_kernel(const float* __restrict__ x4,
                            const float* __restrict__ pw,
                            const float* __restrict__ pb,
                            float* __restrict__ fea) {
  __shared__ float red[8][32][8];
  const int b    = blockIdx.y;
  const int hw0  = blockIdx.x * 32;
  const int lane = threadIdx.x & 31;
  const int wv   = threadIdx.x >> 5;       // 0..7 -> channel chunk
  const int hw   = hw0 + lane;
  const bool ok  = (hw < N_NODES);

  float acc[8];
#pragma unroll
  for (int o = 0; o < 8; ++o) acc[o] = 0.f;

  const int c0 = wv * 256;
  const float* xp = x4 + ((size_t)b * 2048 + c0) * N_NODES + (ok ? hw : 0);
#pragma unroll 4
  for (int c = 0; c < 256; ++c) {
    float xv = ok ? xp[(size_t)c * N_NODES] : 0.f;
#pragma unroll
    for (int o = 0; o < 8; ++o) acc[o] += xv * pw[o * 2048 + c0 + c];
  }
#pragma unroll
  for (int o = 0; o < 8; ++o) red[wv][lane][o] = acc[o];
  __syncthreads();

  if (wv == 0 && ok) {
#pragma unroll
    for (int o = 0; o < 8; ++o) {
      float s = pb[o];
#pragma unroll
      for (int w = 0; w < 8; ++w) s += red[w][lane][o];
      fea[((size_t)b * N_NODES + hw) * 8 + o] = s;
    }
  }
}

__device__ __forceinline__ float leaky02(float x) {
  return x > 0.f ? x : 0.2f * x;
}

// ---------------------------------------------------------------------------
// Stage 2: GAT layer 1 (8 -> 4 heads x 16, concat) + ReLU. One block per batch.
// Grid graph (8-connected + self loop) recomputed; edge_index input ignored.
// ---------------------------------------------------------------------------
__global__ void gat1_kernel(const float* __restrict__ fea,
                            const float* __restrict__ W,      // [8,64]
                            const float* __restrict__ a_src,  // [4,16]
                            const float* __restrict__ a_dst,  // [4,16]
                            const float* __restrict__ bias,   // [64]
                            float* __restrict__ x1) {         // [B,225,64]
  __shared__ float sh_h[N_NODES * 64];   // 57600 B
  __shared__ float sh_as[N_NODES * 4];   //  3600 B
  __shared__ float sh_ad[N_NODES * 4];   //  3600 B
  const int b = blockIdx.x;
  const int t = threadIdx.x;

  // h = fea @ W
  for (int i = t; i < N_NODES * 64; i += 256) {
    const int n = i >> 6, j = i & 63;
    const float* fp = fea + ((size_t)b * N_NODES + n) * 8;
    float s = 0.f;
#pragma unroll
    for (int c = 0; c < 8; ++c) s += fp[c] * W[c * 64 + j];
    sh_h[i] = s;
  }
  __syncthreads();

  // per-node, per-head attention logits
  for (int i = t; i < N_NODES * 4; i += 256) {
    const int n = i >> 2, hd = i & 3;
    float sa = 0.f, sd = 0.f;
#pragma unroll
    for (int d = 0; d < 16; ++d) {
      const float hv = sh_h[n * 64 + hd * 16 + d];
      sa += hv * a_src[hd * 16 + d];
      sd += hv * a_dst[hd * 16 + d];
    }
    sh_as[i] = sa;
    sh_ad[i] = sd;
  }
  __syncthreads();

  // softmax over incoming edges + aggregate
  for (int i = t; i < N_NODES * 4; i += 256) {
    const int n = i >> 2, hd = i & 3;
    const int gi = n / W_G, gj = n % W_G;
    const float ad = sh_ad[i];

    float mx = -1e30f;
    for (int di = -1; di <= 1; ++di)
      for (int dj = -1; dj <= 1; ++dj) {
        const int ni = gi + di, nj = gj + dj;
        if (ni < 0 || ni >= H_G || nj < 0 || nj >= W_G) continue;
        mx = fmaxf(mx, leaky02(sh_as[(ni * W_G + nj) * 4 + hd] + ad));
      }

    float ssum = 0.f;
    float out[16];
#pragma unroll
    for (int d = 0; d < 16; ++d) out[d] = 0.f;
    for (int di = -1; di <= 1; ++di)
      for (int dj = -1; dj <= 1; ++dj) {
        const int ni = gi + di, nj = gj + dj;
        if (ni < 0 || ni >= H_G || nj < 0 || nj >= W_G) continue;
        const int src = ni * W_G + nj;
        const float w = __expf(leaky02(sh_as[src * 4 + hd] + ad) - mx);
        ssum += w;
#pragma unroll
        for (int d = 0; d < 16; ++d) out[d] += w * sh_h[src * 64 + hd * 16 + d];
      }
    const float inv = 1.f / ssum;
#pragma unroll
    for (int d = 0; d < 16; ++d) {
      const float v = out[d] * inv + bias[hd * 16 + d];
      x1[((size_t)b * N_NODES + n) * 64 + hd * 16 + d] = fmaxf(v, 0.f);
    }
  }
}

// ---------------------------------------------------------------------------
// Stage 3: GAT layer 2 (64 -> 1 head x 8, mean==identity) -> flat[B,1800]
// ---------------------------------------------------------------------------
__global__ void gat2_kernel(const float* __restrict__ x1,
                            const float* __restrict__ W,      // [64,8]
                            const float* __restrict__ a_src,  // [1,8]
                            const float* __restrict__ a_dst,  // [1,8]
                            const float* __restrict__ bias,   // [8]
                            float* __restrict__ flat) {       // [B,1800]
  __shared__ float sh_h[N_NODES * 8];
  __shared__ float sh_as[N_NODES];
  __shared__ float sh_ad[N_NODES];
  const int b = blockIdx.x;
  const int t = threadIdx.x;

  for (int i = t; i < N_NODES * 8; i += 256) {
    const int n = i >> 3, j = i & 7;
    const float* xp = x1 + ((size_t)b * N_NODES + n) * 64;
    float s = 0.f;
    for (int c = 0; c < 64; ++c) s += xp[c] * W[c * 8 + j];
    sh_h[i] = s;
  }
  __syncthreads();

  for (int n = t; n < N_NODES; n += 256) {
    float sa = 0.f, sd = 0.f;
#pragma unroll
    for (int d = 0; d < 8; ++d) {
      const float hv = sh_h[n * 8 + d];
      sa += hv * a_src[d];
      sd += hv * a_dst[d];
    }
    sh_as[n] = sa;
    sh_ad[n] = sd;
  }
  __syncthreads();

  for (int n = t; n < N_NODES; n += 256) {
    const int gi = n / W_G, gj = n % W_G;
    const float ad = sh_ad[n];

    float mx = -1e30f;
    for (int di = -1; di <= 1; ++di)
      for (int dj = -1; dj <= 1; ++dj) {
        const int ni = gi + di, nj = gj + dj;
        if (ni < 0 || ni >= H_G || nj < 0 || nj >= W_G) continue;
        mx = fmaxf(mx, leaky02(sh_as[ni * W_G + nj] + ad));
      }

    float ssum = 0.f;
    float out[8];
#pragma unroll
    for (int d = 0; d < 8; ++d) out[d] = 0.f;
    for (int di = -1; di <= 1; ++di)
      for (int dj = -1; dj <= 1; ++dj) {
        const int ni = gi + di, nj = gj + dj;
        if (ni < 0 || ni >= H_G || nj < 0 || nj >= W_G) continue;
        const int src = ni * W_G + nj;
        const float w = __expf(leaky02(sh_as[src] + ad) - mx);
        ssum += w;
#pragma unroll
        for (int d = 0; d < 8; ++d) out[d] += w * sh_h[src * 8 + d];
      }
    const float inv = 1.f / ssum;
#pragma unroll
    for (int d = 0; d < 8; ++d)
      flat[(size_t)b * 1800 + n * 8 + d] = out[d] * inv + bias[d];
  }
}

// ---------------------------------------------------------------------------
// Stages 4/5: WMMA GEMM  C[M,N] = act(A[M,K] @ B[K,N] + bias[N])
// M = 128 (8 tiles of 16). One wave per 16x16 output tile.
// waveId = blockIdx.x*8 + warp; mTile = waveId & 7; nTile = waveId >> 3
// (so all 8 waves of a block share one B column tile -> full L1 reuse of B).
// Main K-loop is guard-free: A via aligned float4 (b128) loads, B via 16
// strided loads at compile-time immediate offsets off one rolling pointer.
// K tail (FC1, K=1800) is branch-free: clamped addresses + cndmask selects.
// fp32 -> fp16 convert on load, fp32 accumulate via v_wmma_f32_16x16x32_f16.
// Fragment layouts per CDNA5 ISA 7.12.2 (wave32).
// ---------------------------------------------------------------------------
__global__ void wmma_gemm_kernel(const float* __restrict__ A,
                                 const float* __restrict__ B,
                                 const float* __restrict__ bias,
                                 float* __restrict__ C,
                                 int N, int K, int relu) {
  const int lane   = threadIdx.x & 31;
  const int waveId = blockIdx.x * (blockDim.x >> 5) + (threadIdx.x >> 5);
  const int mTile  = waveId & 7;          // M=128 -> 8 row tiles (wave-uniform)
  const int nTile  = waveId >> 3;

  const int m0 = mTile << 4;
  const int n0 = nTile << 4;

  const int hi    = lane >> 4;            // 0 or 1 (lane half)
  const int mA    = m0 + (lane & 15);     // A row for this lane
  const int kOffA = hi * 8;               // A: lanes 16-31 start at K+8
  const int nB    = n0 + (lane & 15);     // B column for this lane
  const int kOffB = hi * 8;               // B: lanes 16-31 start at K+8

  const int kFull = K & ~31;

  const float* ap = A + (size_t)mA * K + kOffA;     // +32 per iter
  const float* bp = B + (size_t)kOffB * N + nB;     // +32*N per iter

  v8f acc = {};
  for (int k0 = 0; k0 < kFull; k0 += 32) {
    __builtin_prefetch(bp + (size_t)128 * N, 0, 1); // speculative: OOB dropped

    // A fragment: halves 0..7 -> k = kOffA+0..7, halves 8..15 -> kOffA+16..23
    const float4 a0 = *(const float4*)(ap + 0);
    const float4 a1 = *(const float4*)(ap + 4);
    const float4 a2 = *(const float4*)(ap + 16);
    const float4 a3 = *(const float4*)(ap + 20);
    v16h a;
    a[0]  = (_Float16)a0.x; a[1]  = (_Float16)a0.y;
    a[2]  = (_Float16)a0.z; a[3]  = (_Float16)a0.w;
    a[4]  = (_Float16)a1.x; a[5]  = (_Float16)a1.y;
    a[6]  = (_Float16)a1.z; a[7]  = (_Float16)a1.w;
    a[8]  = (_Float16)a2.x; a[9]  = (_Float16)a2.y;
    a[10] = (_Float16)a2.z; a[11] = (_Float16)a2.w;
    a[12] = (_Float16)a3.x; a[13] = (_Float16)a3.y;
    a[14] = (_Float16)a3.z; a[15] = (_Float16)a3.w;

    // B fragment: halves 0..7 -> k = kOffB+0..7, halves 8..15 -> kOffB+16..23
    v16h b;
#pragma unroll
    for (int i = 0; i < 16; ++i) b[i] = (_Float16)bp[(size_t)(i + ((i >= 8) ? 8 : 0)) * N];

    acc = __builtin_amdgcn_wmma_f32_16x16x32_f16(
        /*neg_a=*/false, a, /*neg_b=*/false, b,
        /*c_mod=*/(short)0, acc, /*reuse_a=*/false, /*reuse_b=*/false);

    ap += 32;
    bp += (size_t)32 * N;
  }

  // K tail (K % 32 != 0): clamped unconditional loads + value selects,
  // no EXEC-mask branches so the wave stays fully convergent for WMMA.
  if (kFull < K) {
    v16h a, b;
#pragma unroll
    for (int i = 0; i < 16; ++i) {
      const int ka = kFull + kOffA + i + ((i >= 8) ? 8 : 0);
      const int kac = (ka < K) ? ka : (K - 1);
      const float av = A[(size_t)mA * K + kac];
      a[i] = (_Float16)((ka < K) ? av : 0.f);

      const int kb  = kFull + kOffB + i + ((i >= 8) ? 8 : 0);
      const int kbc = (kb < K) ? kb : (K - 1);
      const float bv = B[(size_t)kbc * N + nB];
      b[i] = (_Float16)((kb < K) ? bv : 0.f);
    }
    acc = __builtin_amdgcn_wmma_f32_16x16x32_f16(
        false, a, false, b, (short)0, acc, false, false);
  }

  const float bv = bias[n0 + (lane & 15)];
#pragma unroll
  for (int r = 0; r < 8; ++r) {
    const int mOut = m0 + r + hi * 8;     // C/D: VGPR r -> row r (+8 for hi lanes)
    float v = acc[r] + bv;
    if (relu) v = fmaxf(v, 0.f);
    C[(size_t)mOut * N + (n0 + (lane & 15))] = v;
  }
}

// ---------------------------------------------------------------------------
extern "C" void kernel_launch(void* const* d_in, const int* in_sizes, int n_in,
                              void* d_out, int out_size, void* d_ws, size_t ws_size,
                              hipStream_t stream) {
  (void)in_sizes; (void)n_in; (void)out_size; (void)ws_size;

  const float* x4        = (const float*)d_in[0];   // [128,2048,9,25]
  const float* pool_w    = (const float*)d_in[1];   // [8,2048]
  const float* pool_b    = (const float*)d_in[2];   // [8]
  const float* gat1_w    = (const float*)d_in[3];   // [8,64]
  const float* gat1_asrc = (const float*)d_in[4];   // [4,16]
  const float* gat1_adst = (const float*)d_in[5];   // [4,16]
  const float* gat1_b    = (const float*)d_in[6];   // [64]
  const float* gat2_w    = (const float*)d_in[7];   // [64,8]
  const float* gat2_asrc = (const float*)d_in[8];   // [1,8]
  const float* gat2_adst = (const float*)d_in[9];   // [1,8]
  const float* gat2_b    = (const float*)d_in[10];  // [8]
  const float* w1        = (const float*)d_in[11];  // [1800,2048]
  const float* b1        = (const float*)d_in[12];  // [2048]
  const float* w2        = (const float*)d_in[13];  // [2048,20800]
  const float* b2        = (const float*)d_in[14];  // [20800]
  // d_in[15] = edge_index: static 8-connected grid, recomputed on device.

  float* out = (float*)d_out;                       // [128,20800]

  // Workspace carve-up (floats): fea | x1 | flat | hid  (~10.3 MB total,
  // every offset a multiple of 16 floats -> 64B alignment preserved)
  float* fea  = (float*)d_ws;                             // 128*225*8   = 230400
  float* x1   = fea  + (size_t)BATCH * N_NODES * 8;       // 128*225*64  = 1843200
  float* flat = x1   + (size_t)BATCH * N_NODES * 64;      // 128*1800    = 230400
  float* hid  = flat + (size_t)BATCH * 1800;              // 128*2048    = 262144

  // 1) pool: 2048 -> 8 channels
  pool_kernel<<<dim3(8, BATCH), 256, 0, stream>>>(x4, pool_w, pool_b, fea);

  // 2) GAT1 (+ReLU)
  gat1_kernel<<<BATCH, 256, 0, stream>>>(fea, gat1_w, gat1_asrc, gat1_adst, gat1_b, x1);

  // 3) GAT2 -> flat
  gat2_kernel<<<BATCH, 256, 0, stream>>>(x1, gat2_w, gat2_asrc, gat2_adst, gat2_b, flat);

  // 4) FC1: [128,1800] @ [1800,2048] + b1, ReLU -> hid
  wmma_gemm_kernel<<<2048 / 16, 256, 0, stream>>>(flat, w1, b1, hid, 2048, 1800, 1);

  // 5) FC2: [128,2048] @ [2048,20800] + b2 -> out
  wmma_gemm_kernel<<<20800 / 16, 256, 0, stream>>>(hid, w2, b2, out, 20800, 2048, 0);
}
